// VRP_net_12532714570265
// MI455X (gfx1250) — matrix operand order repaired and checked by reading, not compiled, not run on hardware
//
#include <hip/hip_runtime.h>

typedef __attribute__((ext_vector_type(16))) _Float16 v16h;
typedef __attribute__((ext_vector_type(4)))  _Float16 v4h;
typedef __attribute__((ext_vector_type(8)))  float    v8f;

#define cB 128
#define cN 5000
#define cAK 50
#define cSK0 100
#define cSK1 150
#define cMAXSK 150
#define cD 128
#define cFF 512
#define cSteps 8
#define cCAP 50.0f
#define cNEG (-1.0e9f)
#define cBIG (1.0e30f)
#define cGONE (3.0e30f)

// ---------------------------------------------------------------------------
// WMMA fragment helpers (CDNA5 gfx1250, wave32, 16x16x32 f16 -> f32)
// A/B k order: contiguous k = 16*(lane/16) + s, same in both operands (not the ISA table order)
// ---------------------------------------------------------------------------
__device__ __forceinline__ v16h load_a_f16(const _Float16* A, int lda, int m0, int k0) {
    int lane = threadIdx.x & 31;
    const _Float16* row = A + (size_t)(m0 + (lane & 15)) * lda + k0;
    int kb = (lane >> 4) << 4;  // 0 or 16
    v16h a;
#pragma unroll
    for (int v = 0; v < 4; ++v) { a[2*v]   = row[kb + 2*v];      a[2*v+1] = row[kb + 2*v + 1]; }
#pragma unroll
    for (int v = 0; v < 4; ++v) { a[8+2*v] = row[8 + kb + 2*v]; a[9+2*v] = row[8 + kb + 2*v + 1]; }
    return a;
}

__device__ __forceinline__ v16h load_a_f32(const float* A, int lda, int m0, int k0) {
    int lane = threadIdx.x & 31;
    const float* row = A + (size_t)(m0 + (lane & 15)) * lda + k0;
    int kb = (lane >> 4) << 4;
    v16h a;
#pragma unroll
    for (int v = 0; v < 4; ++v) { a[2*v]   = (_Float16)row[kb + 2*v];      a[2*v+1] = (_Float16)row[kb + 2*v + 1]; }
#pragma unroll
    for (int v = 0; v < 4; ++v) { a[8+2*v] = (_Float16)row[8 + kb + 2*v]; a[9+2*v] = (_Float16)row[8 + kb + 2*v + 1]; }
    return a;
}

// B fragment from LDS, row-major K x N (used for P@V where B is the V panel)
__device__ __forceinline__ v16h load_b_f16(const _Float16* Bm, int ldb, int k0, int n0) {
    int lane = threadIdx.x & 31;
    const _Float16* col = Bm + (size_t)(k0 + ((lane >> 4) << 4)) * ldb + n0 + (lane & 15);
    v16h b;
#pragma unroll
    for (int v = 0; v < 8; ++v) { b[2*v] = col[(size_t)(2*v) * ldb]; b[2*v+1] = col[(size_t)(2*v+1) * ldb]; }
    return b;
}

// B fragment from pre-shuffled ("fragment-major") weights: each lane's 16 halves
// are 32 contiguous bytes -> two global_load_b128.
__device__ __forceinline__ v16h load_b_packed(const _Float16* P, int NT, int kt, int nt) {
    return *(const v16h*)(P + (((size_t)kt * NT + nt) * 32 + (threadIdx.x & 31)) * 16);
}

__device__ __forceinline__ v8f load_c_f32(const float* C, int ldc, int m0, int n0) {
    int lane = threadIdx.x & 31;
    const float* p = C + (size_t)(m0 + ((lane >> 4) << 3)) * ldc + n0 + (lane & 15);
    v8f c;
#pragma unroll
    for (int v = 0; v < 8; ++v) c[v] = p[(size_t)v * ldc];
    return c;
}

// Generic in-block GEMM over 16x16 tiles, 8 waves per block (blockDim=256).
// C[M,N] = act(A[M,K] * B[K,N] + bias). BPACK: B in fragment-major global layout
// (NTB = tile columns of full packed matrix, ktOff/ntOff = tile offsets).
// KFIX > 0: compile-time K (fully unrolled WMMA chain).
template <bool AF32, bool OUTF16, bool ACCUM, bool RELU, bool HASBIAS, bool BPACK, int KFIX>
__device__ void wgemm(const void* A, int lda, const _Float16* B, int NTB, int ktOff, int ntOff,
                      const float* bias, void* C, int ldc, int M, int N, int K) {
    const int wid = threadIdx.x >> 5;
    const int nt = N >> 4, mt = M >> 4;
    for (int t = wid; t < mt * nt; t += 8) {
        int m0 = (t / nt) << 4;
        int n0 = (t - (t / nt) * nt) << 4;
        v8f c = {};
        if constexpr (ACCUM) c = load_c_f32((const float*)C, ldc, m0, n0);
        auto kstep = [&](int k0) {
            v16h a;
            if constexpr (AF32) a = load_a_f32((const float*)A, lda, m0, k0);
            else                a = load_a_f16((const _Float16*)A, lda, m0, k0);
            v16h b;
            if constexpr (BPACK) b = load_b_packed(B, NTB, ktOff + (k0 >> 5), ntOff + (n0 >> 4));
            else                 b = load_b_f16(B, NTB, k0, n0);
            c = __builtin_amdgcn_wmma_f32_16x16x32_f16(false, a, false, b, (short)0, c, false, false);
        };
        if constexpr (KFIX > 0) {
#pragma unroll
            for (int k0 = 0; k0 < KFIX; k0 += 32) kstep(k0);
        } else {
            for (int k0 = 0; k0 < K; k0 += 32) kstep(k0);
        }
        int lane = threadIdx.x & 31;
        int n = n0 + (lane & 15);
        int r0 = m0 + ((lane >> 4) << 3);
        float bv = 0.0f;
        if constexpr (HASBIAS) bv = bias[n];
#pragma unroll
        for (int v = 0; v < 8; ++v) {
            float val = c[v];
            if constexpr (HASBIAS) val += bv;
            if constexpr (RELU) val = fmaxf(val, 0.0f);
            if constexpr (OUTF16) ((_Float16*)C)[(size_t)(r0 + v) * ldc + n] = (_Float16)val;
            else                  ((float*)C)[(size_t)(r0 + v) * ldc + n] = val;
        }
    }
}

__device__ __forceinline__ void cvt_block_f32_to_f16(const float* __restrict__ src,
                                                     _Float16* __restrict__ dst, int n) {
    for (int i = threadIdx.x * 4; i < n; i += blockDim.x * 4) {
        float4 f = *(const float4*)(src + i);
        v4h h;
        h[0] = (_Float16)f.x; h[1] = (_Float16)f.y; h[2] = (_Float16)f.z; h[3] = (_Float16)f.w;
        *(v4h*)(dst + i) = h;
    }
}

// ---------------------------------------------------------------------------
// Weight shuffle: row-major f32 [K,N] -> f16 fragment-major
// dst[((kt*NT+nt)*32 + lane)*16 + e] = src[(kt*32 + (lane>>4)*16 + e)*N + nt*16 + (lane&15)]
// ---------------------------------------------------------------------------
__global__ void k_pack_b(const float* __restrict__ src, _Float16* __restrict__ dst, int K, int N) {
    int i = blockIdx.x * blockDim.x + threadIdx.x;
    if (i >= K * N) return;
    int e = i & 15;
    int lane = (i >> 4) & 31;
    int tile = i >> 9;
    int NT = N >> 4;
    int kt = tile / NT, nt = tile - kt * NT;
    int k = kt * 32 + ((lane >> 4) << 4) + e;
    int n = nt * 16 + (lane & 15);
    dst[i] = (_Float16)src[(size_t)k * N + n];
}

__global__ void k_init(const int* __restrict__ demand, const float* __restrict__ depot,
                       float* true_dem, float* last_xy, int* last_idx, float* used, float* logps) {
    int i = blockIdx.x * blockDim.x + threadIdx.x;
    if (i < cB * (cN + 1)) {
        int b = i / (cN + 1), j = i % (cN + 1);
        true_dem[i] = (j < cN) ? (float)demand[b * cN + j] : 0.0f;
    }
    if (i < cB) {
        last_xy[2 * i]     = depot[2 * i];
        last_xy[2 * i + 1] = depot[2 * i + 1];
        last_idx[i] = -1;
        used[i] = 0.0f;
        logps[i] = 0.0f;
    }
}

// ---------------------------------------------------------------------------
// KNN + masks. One block per batch element, 256 threads.
// ---------------------------------------------------------------------------
__global__ void k_knn(const float* __restrict__ nodes, const float* __restrict__ depot,
                      const float* __restrict__ true_dem, const float* __restrict__ last_xy,
                      const int* __restrict__ last_idx, const float* __restrict__ used,
                      float* d2buf, int* a_idx, int* a_mask, int* s_idx, int* s_mask,
                      float* remain_out, int* depot_bsz_out) {
    __shared__ float rv[256];
    __shared__ int   ri[256];
    int b = blockIdx.x, tid = threadIdx.x;
    float rem = (cCAP - used[b]) * (1.0f / cCAP);
    int li = last_idx[b];
    float lx = last_xy[2 * b], ly = last_xy[2 * b + 1];
    float* d2 = d2buf + (size_t)b * cN;

    float dsum = 0.0f;
    for (int j = tid; j < cN; j += 256) {
        __builtin_prefetch(&nodes[((size_t)b * cN + j + 512) * 2], 0, 0);
        float dem = true_dem[b * (cN + 1) + j];
        dsum += dem;
        float dn = dem * (1.0f / cCAP);
        bool avail_excl = !((dn > 0.0f) && (dn < rem));
        bool isl = (j == li) && (li >= 0);
        float dx = nodes[((size_t)b * cN + j) * 2] - lx;
        float dy = nodes[((size_t)b * cN + j) * 2 + 1] - ly;
        d2[j] = (avail_excl || isl) ? cBIG : (dx * dx + dy * dy);
    }
    rv[tid] = dsum;
    __syncthreads();
    for (int s = 128; s > 0; s >>= 1) { if (tid < s) rv[tid] += rv[tid + s]; __syncthreads(); }
    if (tid == 0) {
        remain_out[b] = rem;
        depot_bsz_out[b] = ((li == -1) && (rv[0] > 0.0f)) ? 1 : 0;
    }
    __syncthreads();

    // pass 1: 50 nearest among available
    for (int i = 0; i < cAK; ++i) {
        float best = 3.4e38f; int bj = 1 << 30;
        for (int j = tid; j < cN; j += 256) {
            float v = d2[j];
            if (v < best) { best = v; bj = j; }
        }
        rv[tid] = best; ri[tid] = bj;
        __syncthreads();
        for (int s = 128; s > 0; s >>= 1) {
            if (tid < s) {
                if (rv[tid + s] < rv[tid] || (rv[tid + s] == rv[tid] && ri[tid + s] < ri[tid])) {
                    rv[tid] = rv[tid + s]; ri[tid] = ri[tid + s];
                }
            }
            __syncthreads();
        }
        if (tid == 0) {
            int j = ri[0];
            int m = (rv[0] >= cBIG) ? 1 : 0;
            a_idx[b * cAK + i] = j; a_mask[b * cAK + i] = m;
            s_idx[b * cMAXSK + i] = j; s_mask[b * cMAXSK + i] = m;
            d2[j] = cGONE;
        }
        __syncthreads();
    }

    // pass 2: rebuild with mfs = finished (selected-invalid -> true) | is_last
    for (int j = tid; j < cN; j += 256) {
        float dem = true_dem[b * (cN + 1) + j];
        bool fin = !(dem > 0.0f);
        bool isl = (j == li) && (li >= 0);
        float dx = nodes[((size_t)b * cN + j) * 2] - lx;
        float dy = nodes[((size_t)b * cN + j) * 2 + 1] - ly;
        d2[j] = (fin || isl) ? cBIG : (dx * dx + dy * dy);
    }
    __syncthreads();
    if (tid < cAK) {
        if (a_mask[b * cAK + tid]) d2[a_idx[b * cAK + tid]] = cBIG;
    }
    __syncthreads();
    for (int i = 0; i < (cMAXSK - cAK); ++i) {
        float best = 3.4e38f; int bj = 1 << 30;
        for (int j = tid; j < cN; j += 256) {
            float v = d2[j];
            if (v < best) { best = v; bj = j; }
        }
        rv[tid] = best; ri[tid] = bj;
        __syncthreads();
        for (int s = 128; s > 0; s >>= 1) {
            if (tid < s) {
                if (rv[tid + s] < rv[tid] || (rv[tid + s] == rv[tid] && ri[tid + s] < ri[tid])) {
                    rv[tid] = rv[tid + s]; ri[tid] = ri[tid + s];
                }
            }
            __syncthreads();
        }
        if (tid == 0) {
            int j = ri[0];
            s_idx[b * cMAXSK + cAK + i] = j;
            s_mask[b * cMAXSK + cAK + i] = (rv[0] >= cBIG) ? 1 : 0;
            d2[j] = cGONE;
        }
        __syncthreads();
    }
}

// ---------------------------------------------------------------------------
// Fused transformer encoder: one block per batch element, 256 threads, 8 waves.
// Whole residual stream resident in LDS, all GEMMs via v_wmma_f32_16x16x32_f16.
// ---------------------------------------------------------------------------
struct EncW {
    const _Float16 *Wq, *Wk, *Wv, *Wo, *W1, *W2;             // f16, fragment-major
    const float *W_in, *b_in, *b1, *b2, *ln1s, *ln1b, *ln2s, *ln2b;  // f32
};

__global__ void k_encode(EncW w, int Ktok, int T,
                         const int* __restrict__ idxb, int idxStride, const int* __restrict__ maskb,
                         const float* __restrict__ nodes, const float* __restrict__ depot,
                         const float* __restrict__ true_dem, const float* __restrict__ last_xy,
                         const float* __restrict__ remain,
                         float* __restrict__ emb, int embTokens) {
    extern __shared__ char smem[];
    float*    hf   = (float*)smem;                                      // T x 128 f32
    _Float16* xh   = (_Float16*)(smem + (size_t)T * 512);               // T x 128 f16
    float*    oacc = (float*)(smem + (size_t)T * 768);                  // T x 128 f32
    _Float16* qh   = (_Float16*)(smem + (size_t)T * 1280);              // T x 16 f16
    _Float16* kh   = qh + T * 16;
    _Float16* vh   = kh + T * 16;
    float*    sc   = (float*)(smem + (size_t)T * 1376);                 // T x T f32 (also f16 FF buf)
    int*      km   = (int*)((char*)sc + (size_t)T * T * 4);             // T ints

    int b = blockIdx.x;
    float rem = remain[b];
    float lx = last_xy[2 * b], ly = last_xy[2 * b + 1];
    float dpx = depot[2 * b], dpy = depot[2 * b + 1];

    // feats -> W_in projection (3-wide inner dim: VALU)
    for (int t = threadIdx.x; t < T; t += blockDim.x) {
        float f0, f1, f2; int m;
        if (t < Ktok) {
            int j = idxb[b * idxStride + t];
            m = maskb[b * idxStride + t];
            if (j < 0 || j >= cN) j = 0;
            f0 = nodes[((size_t)b * cN + j) * 2];
            f1 = nodes[((size_t)b * cN + j) * 2 + 1];
            f2 = m ? 0.0f : true_dem[b * (cN + 1) + j] * (1.0f / cCAP);
        } else if (t == Ktok)     { f0 = lx;  f1 = ly;  f2 = rem; m = 0; }
        else if (t == Ktok + 1)   { f0 = dpx; f1 = dpy; f2 = rem; m = 0; }
        else                      { f0 = f1 = f2 = 0.0f; m = 1; }
        km[t] = m;
        for (int d = 0; d < 128; ++d)
            hf[t * 128 + d] = f0 * w.W_in[d] + f1 * w.W_in[128 + d] + f2 * w.W_in[256 + d] + w.b_in[d];
    }
    __syncthreads();

    for (int l = 0; l < 2; ++l) {
        // ---- MHA ----
        cvt_block_f32_to_f16(hf, xh, T * 128);
        __syncthreads();
        for (int hh = 0; hh < 8; ++hh) {
            wgemm<false, true, false, false, false, true, 128>(xh, 128, w.Wq + l * 16384, 8, 0, hh, nullptr, qh, 16, T, 16, 128);
            wgemm<false, true, false, false, false, true, 128>(xh, 128, w.Wk + l * 16384, 8, 0, hh, nullptr, kh, 16, T, 16, 128);
            wgemm<false, true, false, false, false, true, 128>(xh, 128, w.Wv + l * 16384, 8, 0, hh, nullptr, vh, 16, T, 16, 128);
            __syncthreads();
            // scores (DH=16 -> VALU), masked
            for (int p = threadIdx.x; p < T * T; p += blockDim.x) {
                int i = p / T, j = p - i * T;
                float s;
                if (km[j]) s = cNEG;
                else {
                    s = 0.0f;
#pragma unroll
                    for (int d = 0; d < 16; ++d) s += (float)qh[i * 16 + d] * (float)kh[j * 16 + d];
                    s *= 0.25f;  // 1/sqrt(16)
                }
                sc[p] = s;
            }
            __syncthreads();
            // row softmax
            for (int i = threadIdx.x; i < T; i += blockDim.x) {
                float mx = -3.4e38f;
                for (int j = 0; j < T; ++j) mx = fmaxf(mx, sc[i * T + j]);
                float sum = 0.0f;
                for (int j = 0; j < T; ++j) { float e = __expf(sc[i * T + j] - mx); sc[i * T + j] = e; sum += e; }
                float inv = 1.0f / sum;
                for (int j = 0; j < T; ++j) sc[i * T + j] *= inv;
            }
            __syncthreads();
            // o_h = P @ V  (WMMA, A f32->f16 on load), into oacc columns [hh*16, hh*16+16)
            wgemm<true, false, false, false, false, false, 0>(sc, T, vh, 16, 0, 0, nullptr, oacc + hh * 16, 128, T, 16, T);
            __syncthreads();
        }
        // attn_out @ Wo
        cvt_block_f32_to_f16(oacc, xh, T * 128);
        __syncthreads();
        wgemm<false, false, false, false, false, true, 128>(xh, 128, w.Wo + l * 16384, 8, 0, 0, nullptr, oacc, 128, T, 128, 128);
        __syncthreads();
        // LN1 over residual
        for (int t = threadIdx.x; t < T; t += blockDim.x) {
            float mean = 0.0f;
            for (int d = 0; d < 128; ++d) mean += hf[t * 128 + d] + oacc[t * 128 + d];
            mean *= (1.0f / 128.0f);
            float var = 0.0f;
            for (int d = 0; d < 128; ++d) {
                float dv = hf[t * 128 + d] + oacc[t * 128 + d] - mean;
                var += dv * dv;
            }
            var *= (1.0f / 128.0f);
            float inv = rsqrtf(var + 1e-5f);
            for (int d = 0; d < 128; ++d) {
                float v = (hf[t * 128 + d] + oacc[t * 128 + d] - mean) * inv;
                hf[t * 128 + d] = v * w.ln1s[l * 128 + d] + w.ln1b[l * 128 + d];
            }
        }
        __syncthreads();
        // ---- FF (chunked over FF=512 in 4 x 128 columns) ----
        cvt_block_f32_to_f16(hf, xh, T * 128);
        __syncthreads();
        for (int i = threadIdx.x; i < T * 128; i += blockDim.x) oacc[i] = w.b2[l * 128 + (i & 127)];
        __syncthreads();
        _Float16* y1 = (_Float16*)sc;
        for (int c = 0; c < 4; ++c) {
            wgemm<false, true, false, true, true, true, 128>(xh, 128, w.W1 + l * 65536, 32, 0, c * 8,
                                                             w.b1 + l * 512 + c * 128, y1, 128, T, 128, 128);
            __syncthreads();
            wgemm<false, false, true, false, false, true, 128>(y1, 128, w.W2 + l * 65536, 8, c * 4, 0,
                                                               nullptr, oacc, 128, T, 128, 128);
            __syncthreads();
        }
        // LN2 over residual
        for (int t = threadIdx.x; t < T; t += blockDim.x) {
            float mean = 0.0f;
            for (int d = 0; d < 128; ++d) mean += hf[t * 128 + d] + oacc[t * 128 + d];
            mean *= (1.0f / 128.0f);
            float var = 0.0f;
            for (int d = 0; d < 128; ++d) {
                float dv = hf[t * 128 + d] + oacc[t * 128 + d] - mean;
                var += dv * dv;
            }
            var *= (1.0f / 128.0f);
            float inv = rsqrtf(var + 1e-5f);
            for (int d = 0; d < 128; ++d) {
                float v = (hf[t * 128 + d] + oacc[t * 128 + d] - mean) * inv;
                hf[t * 128 + d] = v * w.ln2s[l * 128 + d] + w.ln2b[l * 128 + d];
            }
        }
        __syncthreads();
    }
    // write out first Ktok+2 tokens (vectorized)
    {
        float* dst = emb + (size_t)b * embTokens * 128;
        int n = (Ktok + 2) * 128;
        for (int i = threadIdx.x * 4; i < n; i += blockDim.x * 4)
            *(float4*)(dst + i) = *(const float4*)(hf + i);
    }
}

// ---------------------------------------------------------------------------
// Gather/concat kernels for decoder inputs
// ---------------------------------------------------------------------------
__global__ void k_build_q(const float* __restrict__ ea, const float* __restrict__ es0,
                          const float* __restrict__ es1, float* __restrict__ q) {
    int i = blockIdx.x * blockDim.x + threadIdx.x;
    if (i >= cB * 768) return;
    int b = i / 768, c = i - b * 768, seg = c >> 7, d = c & 127;
    float v;
    switch (seg) {
        case 0: v = ea[((size_t)b * 52 + 50) * 128 + d]; break;
        case 1: v = ea[((size_t)b * 52 + 51) * 128 + d]; break;
        case 2: v = es0[((size_t)b * 102 + 100) * 128 + d]; break;
        case 3: v = es0[((size_t)b * 102 + 101) * 128 + d]; break;
        case 4: v = es1[((size_t)b * 152 + 150) * 128 + d]; break;
        default: v = es1[((size_t)b * 152 + 151) * 128 + d]; break;
    }
    q[i] = v;
}

__global__ void k_build_o(const float* __restrict__ ea, const float* __restrict__ es0,
                          const float* __restrict__ es1, float* __restrict__ o) {
    int i = blockIdx.x * blockDim.x + threadIdx.x;
    if (i >= cB * 51 * 384) return;
    int b = i / (51 * 384);
    int r = i - b * (51 * 384);
    int t = r / 384, c = r - t * 384;
    int enc = c >> 7, d = c & 127;
    float v;
    if (enc == 0)      v = ea[((size_t)b * 52  + (t < 50 ? t : 51))  * 128 + d];
    else if (enc == 1) v = es0[((size_t)b * 102 + (t < 50 ? t : 101)) * 128 + d];
    else               v = es1[((size_t)b * 152 + (t < 50 ? t : 151)) * 128 + d];
    o[i] = v;
}

// Global GEMM: one wave per 16x16 output tile. C = A(f32) * Bpacked(f16) + bias.
__global__ void k_gemm(const float* __restrict__ A, int lda, const _Float16* __restrict__ W, int NTB,
                       const float* __restrict__ bias, float* __restrict__ C, int ldc, int K) {
    int m0 = blockIdx.x << 4, n0 = blockIdx.y << 4;
    v8f c = {};
#pragma unroll 4
    for (int k0 = 0; k0 < K; k0 += 32) {
        v16h a = load_a_f32(A, lda, m0, k0);
        v16h b = load_b_packed(W, NTB, k0 >> 5, blockIdx.y);
        c = __builtin_amdgcn_wmma_f32_16x16x32_f16(false, a, false, b, (short)0, c, false, false);
    }
    int lane = threadIdx.x & 31;
    int n = n0 + (lane & 15);
    int r0 = m0 + ((lane >> 4) << 3);
    float bv = bias[n];
#pragma unroll
    for (int v = 0; v < 8; ++v) C[(size_t)(r0 + v) * ldc + n] = c[v] + bv;
}

// ---------------------------------------------------------------------------
// Decoder + argmax + state update. One block (128 threads) per batch element.
// ---------------------------------------------------------------------------
__global__ void k_decode(const float* __restrict__ h_q, const float* __restrict__ Kd,
                         const float* __restrict__ Vd, const int* __restrict__ a_idx,
                         const int* __restrict__ a_mask, const int* __restrict__ depot_bsz,
                         const float* __restrict__ dWq, const float* __restrict__ dWo,
                         const float* __restrict__ nodes, const float* __restrict__ depot,
                         float* true_dem, float* last_xy, int* last_idx, float* used,
                         float* logps, float* out, int step) {
    __shared__ float h[128], qv[128], att[128], tmp[128], sc2[64], pr[64];
    int b = blockIdx.x, tid = threadIdx.x;
    h[tid] = h_q[b * 128 + tid];
    __syncthreads();

    for (int l = 0; l < 2; ++l) {
        float s = 0.0f;
        for (int k = 0; k < 128; ++k) s += h[k] * dWq[(l * 128 + k) * 128 + tid];
        qv[tid] = s;
        __syncthreads();
        for (int hh = 0; hh < 8; ++hh) {
            if (tid < 51) {
                const float* kp = Kd + ((size_t)b * 51 + tid) * 384 + l * 128 + hh * 16;
                float sdot = 0.0f;
#pragma unroll
                for (int d = 0; d < 16; ++d) sdot += qv[hh * 16 + d] * kp[d];
                sdot *= 0.25f;
                int m = (tid < 50) ? a_mask[b * cAK + tid] : depot_bsz[b];
                sc2[tid] = m ? cNEG : sdot;
            }
            __syncthreads();
            if (tid == 0) {
                float mx = -3.4e38f;
                for (int j = 0; j < 51; ++j) mx = fmaxf(mx, sc2[j]);
                float sum = 0.0f;
                for (int j = 0; j < 51; ++j) { float e = __expf(sc2[j] - mx); pr[j] = e; sum += e; }
                float inv = 1.0f / sum;
                for (int j = 0; j < 51; ++j) pr[j] *= inv;
            }
            __syncthreads();
            if (tid < 16) {
                float o = 0.0f;
                for (int j = 0; j < 51; ++j)
                    o += pr[j] * Vd[((size_t)b * 51 + j) * 384 + l * 128 + hh * 16 + tid];
                att[hh * 16 + tid] = o;
            }
            __syncthreads();
        }
        float so = 0.0f;
        for (int k = 0; k < 128; ++k) so += att[k] * dWo[(l * 128 + k) * 128 + tid];
        tmp[tid] = so;
        __syncthreads();
        h[tid] += tmp[tid];
        __syncthreads();
    }

    // final pointer logits
    float s = 0.0f;
    for (int k = 0; k < 128; ++k) s += h[k] * dWq[(2 * 128 + k) * 128 + tid];
    qv[tid] = s;
    __syncthreads();
    if (tid < 51) {
        const float* kp = Kd + ((size_t)b * 51 + tid) * 384 + 256;
        float sdot = 0.0f;
        for (int d = 0; d < 128; ++d) sdot += qv[d] * kp[d];
        float lg = 10.0f * tanhf(sdot * 0.08838834764831845f);  // 1/sqrt(128)
        int m = (tid < 50) ? a_mask[b * cAK + tid] : depot_bsz[b];
        sc2[tid] = m ? cNEG : lg;
    }
    __syncthreads();
    if (tid == 0) {
        float mx = -3.4e38f;
        for (int j = 0; j < 51; ++j) mx = fmaxf(mx, sc2[j]);
        float sum = 0.0f;
        for (int j = 0; j < 51; ++j) { float e = __expf(sc2[j] - mx); pr[j] = e; sum += e; }
        float inv = 1.0f / sum;
        int choice = 0; float best = -1.0f;
        for (int j = 0; j < 51; ++j) { pr[j] *= inv; if (pr[j] > best) { best = pr[j]; choice = j; } }
        int nxt = (choice < 50) ? a_idx[b * cAK + choice] : -1;
        int w = (nxt < 0) ? cN : nxt;
        float nlx = (w == cN) ? depot[2 * b]     : nodes[((size_t)b * cN + w) * 2];
        float nly = (w == cN) ? depot[2 * b + 1] : nodes[((size_t)b * cN + w) * 2 + 1];
        float td = true_dem[b * (cN + 1) + w];
        float nu = (td + used[b]) * ((nxt != -1) ? 1.0f : 0.0f);
        true_dem[b * (cN + 1) + w] = 0.0f;
        used[b] = nu;
        last_idx[b] = nxt;
        last_xy[2 * b] = nlx;
        last_xy[2 * b + 1] = nly;
        float lp = __logf(fmaxf(pr[choice], 1e-30f));
        float acc = ((step == 0) ? 0.0f : logps[b]) + lp;
        logps[b] = acc;
        out[b * cSteps + step] = (float)nxt;
        if (step == cSteps - 1) out[cB * cSteps + b] = acc;
    }
}

// ---------------------------------------------------------------------------
// Host orchestration
// ---------------------------------------------------------------------------
static size_t enc_lds(int T) {
    return (size_t)T * 1376 + (size_t)T * T * 4 + (size_t)T * 4;
}

extern "C" void kernel_launch(void* const* d_in, const int* in_sizes, int n_in,
                              void* d_out, int out_size, void* d_ws, size_t ws_size,
                              hipStream_t stream) {
    (void)in_sizes; (void)n_in; (void)out_size; (void)ws_size;
    // d_in layout: setup_inputs() dict insertion order.
    // 0 nodes, 1 depot, 2 demand; per-encoder leaves in _mk_encoder insertion order:
    // W_in,b_in,Wq,Wk,Wv,Wo,W1,b1,W2,b2,ln1_s,ln1_b,ln2_s,ln2_b
    // action: 3..16 ; state0: 17..30 ; state1: 31..44 ;
    // 45 Wq_mlp, 46 bq_mlp, 47 WK, 48 bK, 49 WV, 50 bV, 51 dec_Wq, 52 dec_Wo
    const float* nodes  = (const float*)d_in[0];
    const float* depot  = (const float*)d_in[1];
    const int*   demand = (const int*)d_in[2];
    const int encBase[3] = {3, 17, 31};
    const float* Wq_mlp = (const float*)d_in[45];
    const float* bq_mlp = (const float*)d_in[46];
    const float* WKf    = (const float*)d_in[47];
    const float* bK     = (const float*)d_in[48];
    const float* WVf    = (const float*)d_in[49];
    const float* bV     = (const float*)d_in[50];
    const float* dWq    = (const float*)d_in[51];
    const float* dWo    = (const float*)d_in[52];

    char* ws = (char*)d_ws;
    size_t off = 0;
    auto take = [&](size_t bytes) -> char* {
        char* p = ws + off;
        off = (off + bytes + 255) & ~(size_t)255;
        return p;
    };

    // f16 fragment-major weight pools
    _Float16* encWq16[3]; _Float16* encWk16[3]; _Float16* encWv16[3];
    _Float16* encWo16[3]; _Float16* encW1_16[3]; _Float16* encW2_16[3];
    for (int e = 0; e < 3; ++e) {
        encWq16[e]  = (_Float16*)take((size_t)2 * 128 * 128 * 2);
        encWk16[e]  = (_Float16*)take((size_t)2 * 128 * 128 * 2);
        encWv16[e]  = (_Float16*)take((size_t)2 * 128 * 128 * 2);
        encWo16[e]  = (_Float16*)take((size_t)2 * 128 * 128 * 2);
        encW1_16[e] = (_Float16*)take((size_t)2 * 128 * 512 * 2);
        encW2_16[e] = (_Float16*)take((size_t)2 * 512 * 128 * 2);
    }
    _Float16* Wq_mlp16 = (_Float16*)take((size_t)768 * 128 * 2);
    _Float16* WK16     = (_Float16*)take((size_t)384 * 384 * 2);
    _Float16* WV16     = (_Float16*)take((size_t)384 * 384 * 2);

    // state + scratch buffers
    float* true_dem = (float*)take((size_t)cB * (cN + 1) * 4);
    float* last_xy  = (float*)take((size_t)cB * 2 * 4);
    int*   last_idx = (int*)take((size_t)cB * 4);
    float* used     = (float*)take((size_t)cB * 4);
    float* logps    = (float*)take((size_t)cB * 4);
    float* remain   = (float*)take((size_t)cB * 4);
    int*   depot_bz = (int*)take((size_t)cB * 4);
    float* d2buf    = (float*)take((size_t)cB * cN * 4);
    int*   a_idx    = (int*)take((size_t)cB * cAK * 4);
    int*   a_mask   = (int*)take((size_t)cB * cAK * 4);
    int*   s_idx    = (int*)take((size_t)cB * cMAXSK * 4);
    int*   s_mask   = (int*)take((size_t)cB * cMAXSK * 4);
    float* emb_a    = (float*)take((size_t)cB * 52 * 128 * 4);
    float* emb_s0   = (float*)take((size_t)cB * 102 * 128 * 4);
    float* emb_s1   = (float*)take((size_t)cB * 152 * 128 * 4);
    float* emb_q    = (float*)take((size_t)cB * 768 * 4);
    float* emb_o    = (float*)take((size_t)cB * 51 * 384 * 4);
    float* h_q      = (float*)take((size_t)cB * 128 * 4);
    float* Kd       = (float*)take((size_t)cB * 51 * 384 * 4);
    float* Vd       = (float*)take((size_t)cB * 51 * 384 * 4);

    auto pack = [&](const float* s, _Float16* d, int K, int N) {
        int n = K * N;
        k_pack_b<<<(n + 255) / 256, 256, 0, stream>>>(s, d, K, N);
    };
    for (int e = 0; e < 3; ++e) {
        int bse = encBase[e];
        for (int l = 0; l < 2; ++l) {
            pack((const float*)d_in[bse + 2] + l * 16384, encWq16[e] + l * 16384, 128, 128);
            pack((const float*)d_in[bse + 3] + l * 16384, encWk16[e] + l * 16384, 128, 128);
            pack((const float*)d_in[bse + 4] + l * 16384, encWv16[e] + l * 16384, 128, 128);
            pack((const float*)d_in[bse + 5] + l * 16384, encWo16[e] + l * 16384, 128, 128);
            pack((const float*)d_in[bse + 6] + l * 65536, encW1_16[e] + l * 65536, 128, 512);
            pack((const float*)d_in[bse + 8] + l * 65536, encW2_16[e] + l * 65536, 512, 128);
        }
    }
    pack(Wq_mlp, Wq_mlp16, 768, 128);
    pack(WKf, WK16, 384, 384);
    pack(WVf, WV16, 384, 384);

    k_init<<<(cB * (cN + 1) + 255) / 256, 256, 0, stream>>>(
        demand, depot, true_dem, last_xy, last_idx, used, logps);

    EncW encw[3];
    for (int e = 0; e < 3; ++e) {
        int bse = encBase[e];
        encw[e].Wq = encWq16[e]; encw[e].Wk = encWk16[e]; encw[e].Wv = encWv16[e];
        encw[e].Wo = encWo16[e]; encw[e].W1 = encW1_16[e]; encw[e].W2 = encW2_16[e];
        encw[e].W_in = (const float*)d_in[bse + 0];
        encw[e].b_in = (const float*)d_in[bse + 1];
        encw[e].b1   = (const float*)d_in[bse + 7];
        encw[e].b2   = (const float*)d_in[bse + 9];
        encw[e].ln1s = (const float*)d_in[bse + 10];
        encw[e].ln1b = (const float*)d_in[bse + 11];
        encw[e].ln2s = (const float*)d_in[bse + 12];
        encw[e].ln2b = (const float*)d_in[bse + 13];
    }

    const size_t ldsA = enc_lds(64), ldsS0 = enc_lds(128), ldsS1 = enc_lds(160);

    for (int step = 0; step < cSteps; ++step) {
        k_knn<<<cB, 256, 0, stream>>>(nodes, depot, true_dem, last_xy, last_idx, used,
                                      d2buf, a_idx, a_mask, s_idx, s_mask, remain, depot_bz);
        k_encode<<<cB, 256, ldsA, stream>>>(encw[0], cAK, 64, a_idx, cAK, a_mask,
                                            nodes, depot, true_dem, last_xy, remain, emb_a, 52);
        k_encode<<<cB, 256, ldsS0, stream>>>(encw[1], cSK0, 128, s_idx, cMAXSK, s_mask,
                                             nodes, depot, true_dem, last_xy, remain, emb_s0, 102);
        k_encode<<<cB, 256, ldsS1, stream>>>(encw[2], cSK1, 160, s_idx, cMAXSK, s_mask,
                                             nodes, depot, true_dem, last_xy, remain, emb_s1, 152);
        k_build_q<<<(cB * 768 + 255) / 256, 256, 0, stream>>>(emb_a, emb_s0, emb_s1, emb_q);
        k_build_o<<<(cB * 51 * 384 + 255) / 256, 256, 0, stream>>>(emb_a, emb_s0, emb_s1, emb_o);
        k_gemm<<<dim3(cB / 16, 128 / 16), 32, 0, stream>>>(emb_q, 768, Wq_mlp16, 8, bq_mlp, h_q, 128, 768);
        k_gemm<<<dim3(cB * 51 / 16, 384 / 16), 32, 0, stream>>>(emb_o, 384, WK16, 24, bK, Kd, 384, 384);
        k_gemm<<<dim3(cB * 51 / 16, 384 / 16), 32, 0, stream>>>(emb_o, 384, WV16, 24, bV, Vd, 384, 384);
        k_decode<<<cB, 128, 0, stream>>>(h_q, Kd, Vd, a_idx, a_mask, depot_bz, dWq, dWo,
                                         nodes, depot, true_dem, last_xy, last_idx, used,
                                         logps, (float*)d_out, step);
    }
}
